// DependencyParser_47949014892843
// MI455X (gfx1250) — matrix operand order repaired and checked by reading, not compiled, not run on hardware
//
#include <hip/hip_runtime.h>
#include <math.h>

// ---- problem constants ----
#define SEQ   512
#define WD    300
#define TD    100
#define CEMB  20
#define CHID  20
#define HID   512
#define LABN  50
#define MAXW  16
#define KE    416   // WD+TD=400 padded to 13*32
#define KA    544   // 513 padded to 17*32 (K for biaffine GEMMs)
#define NLP   528   // 513 padded to 33*16 (N for label biaffine)
#define NTL   33    // NLP/16

typedef __attribute__((ext_vector_type(16))) __bf16    v16bf;
typedef __attribute__((ext_vector_type(8)))  float     v8f;
typedef __attribute__((ext_vector_type(4)))  unsigned  uv4;
typedef __attribute__((ext_vector_type(4)))  float     fv4;

union Frag16 { unsigned u[8]; uv4 q[2]; v16bf v; };

__device__ __forceinline__ unsigned short f2bf(float f) {
  unsigned u = __float_as_uint(f);
  u += 0x7fffu + ((u >> 16) & 1u);          // RNE
  return (unsigned short)(u >> 16);
}
__device__ __forceinline__ float bflo(unsigned p) { return __uint_as_float(p << 16); }
__device__ __forceinline__ float bfhi(unsigned p) { return __uint_as_float(p & 0xffff0000u); }
__device__ __forceinline__ float sigm(float x) { return 1.0f / (1.0f + __expf(-x)); }

// ---------------------------------------------------------------------------
// Embedding gather -> bf16 A matrices (normal + time-reversed), zero padded.
// ---------------------------------------------------------------------------
__global__ void k_embed(const int* __restrict__ sent, const int* __restrict__ tags,
                        const float* __restrict__ wemb, const float* __restrict__ temb,
                        unsigned short* __restrict__ embA, unsigned short* __restrict__ embR) {
  int idx = blockIdx.x * blockDim.x + threadIdx.x;
  if (idx >= SEQ * KE) return;
  int s = idx / KE, j = idx % KE;
  float v = 0.0f;
  if (j < WD)            v = wemb[(size_t)sent[s] * WD + j];
  else if (j < WD + TD)  v = temb[(size_t)tags[s] * TD + (j - WD)];
  unsigned short b = f2bf(v);
  embA[s * KE + j] = b;
  embR[(SEQ - 1 - s) * KE + j] = b;
}

// ---------------------------------------------------------------------------
// Pack f32 src[N,K] (row-major, true dims srcN x srcK) into the WMMA
// B-fragment-major layout:  dst_u32[((k/32)*Np + n)*16 + ((k/2)&15)]
// so each lane's 8 B dwords are two 16B chunks, 32B apart (2x global_load_b128).
// Zero-pads to Kp x Np.  grid.z batches with element strides.
// ---------------------------------------------------------------------------
__global__ void k_packB(const float* __restrict__ src, int srcN, int srcK,
                        unsigned* __restrict__ dst, int Kp, int Np,
                        long srcBatch, long dstBatch) {
  src += (size_t)blockIdx.z * srcBatch;
  dst += (size_t)blockIdx.z * dstBatch;
  int idx = blockIdx.x * blockDim.x + threadIdx.x;
  int total = (Kp >> 1) * Np;
  if (idx >= total) return;
  int pairI = idx / Np, n = idx % Np;
  int k0 = pairI * 2;
  float f0 = (n < srcN && k0     < srcK) ? src[(size_t)n * srcK + k0]     : 0.0f;
  float f1 = (n < srcN && k0 + 1 < srcK) ? src[(size_t)n * srcK + k0 + 1] : 0.0f;
  int kt = pairI >> 4, p = pairI & 15;
  dst[((size_t)kt * Np + n) * 16 + p] = (unsigned)f2bf(f0) | ((unsigned)f2bf(f1) << 16);
}

// plain row-major f32 -> bf16 convert (for W_hh used by the recurrent kernel)
__global__ void k_convert(const float* __restrict__ src, unsigned short* __restrict__ dst, int n) {
  int i = blockIdx.x * blockDim.x + threadIdx.x;
  if (i < n) dst[i] = f2bf(src[i]);
}

// Fill the augmentation columns (512..543) of the "_a" bf16 buffers: col512=1, rest 0.
__global__ void k_init_aug(unsigned short* __restrict__ a0, unsigned short* __restrict__ a1) {
  int idx = blockIdx.x * blockDim.x + threadIdx.x;
  if (idx >= 2 * SEQ * 32) return;
  unsigned short* buf = (idx < SEQ * 32) ? a0 : a1;
  int rem = idx % (SEQ * 32);
  int s = rem / 32, n = HID + (rem % 32);
  buf[s * KA + n] = (n == HID) ? (unsigned short)0x3f80 : (unsigned short)0;
}

// ---------------------------------------------------------------------------
// bf16 WMMA GEMM: C[M,N] = A[M,Kp] * B[Kp,N] (+bias) (relu).
// A: pair-interleaved row-major bf16; B: fragment-major (k_packB).
// Each wave computes TWO 16x16 M-tiles sharing one B fragment (2 WMMA/k-step,
// halves B traffic). 4 waves/block along N. All loads are 128-bit.
// Compile-time epilogue selection (f32 out / bf16 out / bias / relu).
// ---------------------------------------------------------------------------
template<bool HASF, bool HASH, bool HASB, bool RELU>
__global__ void k_gemm(const unsigned* __restrict__ A, int Kp,
                       const unsigned* __restrict__ B, int N,
                       float* __restrict__ Cf, int ldc,
                       unsigned short* __restrict__ Ch, int ldh,
                       const float* __restrict__ bias) {
  int lane  = threadIdx.x & 31;
  int wave  = threadIdx.x >> 5;
  int ntile = blockIdx.x * 4 + wave;
  if (ntile >= (N >> 4)) return;
  int mbase = blockIdx.y << 5;              // two 16-row tiles
  int n15 = lane & 15, hi = lane >> 4;
  int n = (ntile << 4) + n15;
  int Kp2 = Kp >> 1;
  const uv4* A4 = (const uv4*)A;
  const uv4* B4 = (const uv4*)B;
  size_t arow0 = (size_t)(mbase + n15) * Kp2;
  size_t arow1 = (size_t)(mbase + 16 + n15) * Kp2;
  v8f c0 = {}, c1 = {};
  for (int kb = 0; kb < Kp; kb += 32) {
    int ak = (kb >> 1) + 4 * hi;            // multiple of 4 dwords -> 16B aligned
    Frag16 a0, a1, b;
    size_t i0 = (arow0 + ak) >> 2, i1 = (arow1 + ak) >> 2;
    a0.q[0] = A4[i0];      a0.q[1] = A4[i0 + 2];
    a1.q[0] = A4[i1];      a1.q[1] = A4[i1 + 2];
    size_t bb = ((size_t)(kb >> 5) * N + n) * 4 + hi;
    b.q[0] = B4[bb];       b.q[1] = B4[bb + 2];
    c0 = __builtin_amdgcn_wmma_f32_16x16x32_bf16(false, a0.v, false, b.v,
                                                 (short)0, c0, false, false);
    c1 = __builtin_amdgcn_wmma_f32_16x16x32_bf16(false, a1.v, false, b.v,
                                                 (short)0, c1, false, false);
  }
  float bv = HASB ? bias[n] : 0.0f;
#pragma unroll
  for (int r = 0; r < 8; ++r) {
    float x0 = c0[r] + bv, x1 = c1[r] + bv;
    if (RELU) { x0 = fmaxf(x0, 0.0f); x1 = fmaxf(x1, 0.0f); }
    int m0 = mbase + r + 8 * hi, m1 = m0 + 16;
    if (HASF) { Cf[(size_t)m0 * ldc + n] = x0; Cf[(size_t)m1 * ldc + n] = x1; }
    if (HASH) { Ch[(size_t)m0 * ldh + n] = f2bf(x0); Ch[(size_t)m1 * ldh + n] = f2bf(x1); }
  }
}

// ---------------------------------------------------------------------------
// Recurrent BiLSTM: one 1024-thread block per direction, resident on a WGP.
// z_x = X@W_ih.T + b precomputed (ZX). Per step: 2048x512 matvec (bf16 W_hh
// streamed 128-bit from L2, 2 rows/lane), h shared via LDS (128-bit reads).
// ---------------------------------------------------------------------------
__global__ void __launch_bounds__(1024, 1)
k_lstm(const float* __restrict__ ZXf, const float* __restrict__ ZXb,
       const unsigned short* __restrict__ WhhF, const unsigned short* __restrict__ WhhB,
       unsigned short* __restrict__ out /* bf16 lstm_out [SEQ][2*HID] */) {
  __shared__ float hsh[HID];
  __shared__ float zrec[4 * HID];
  int tid = threadIdx.x;
  int dir = blockIdx.x;
  const float* ZX  = dir ? ZXb : ZXf;
  const uv4*   Whh = (const uv4*)(dir ? WhhB : WhhF);
  if (tid < HID) hsh[tid] = 0.0f;
  float creg = 0.0f;
  __syncthreads();
  const uv4* w0 = Whh + (size_t)tid * (HID / 8);          // 64 x 16B per row
  const uv4* w1 = Whh + (size_t)(tid + 1024) * (HID / 8);
  const fv4* h4 = (const fv4*)hsh;
  for (int t = 0; t < SEQ; ++t) {
    float a0 = 0.0f, a1 = 0.0f;
    for (int i = 0; i < HID / 8; ++i) {
      uv4 p0 = w0[i], p1 = w1[i];
      fv4 ha = h4[2 * i], hb = h4[2 * i + 1];
#pragma unroll
      for (int j = 0; j < 2; ++j) {
        a0 = fmaf(bflo(p0[j]), ha[2 * j], a0); a0 = fmaf(bfhi(p0[j]), ha[2 * j + 1], a0);
        a1 = fmaf(bflo(p1[j]), ha[2 * j], a1); a1 = fmaf(bfhi(p1[j]), ha[2 * j + 1], a1);
        a0 = fmaf(bflo(p0[2 + j]), hb[2 * j], a0); a0 = fmaf(bfhi(p0[2 + j]), hb[2 * j + 1], a0);
        a1 = fmaf(bflo(p1[2 + j]), hb[2 * j], a1); a1 = fmaf(bfhi(p1[2 + j]), hb[2 * j + 1], a1);
      }
    }
    zrec[tid] = a0; zrec[tid + 1024] = a1;
    __syncthreads();
    if (tid < HID) {
      const float* zx = ZX + (size_t)t * (4 * HID);
      float zi = zx[tid]           + zrec[tid];
      float zf = zx[HID + tid]     + zrec[HID + tid];
      float zg = zx[2 * HID + tid] + zrec[2 * HID + tid];
      float zo = zx[3 * HID + tid] + zrec[3 * HID + tid];
      creg = sigm(zf) * creg + sigm(zi) * tanhf(zg);
      float h = sigm(zo) * tanhf(creg);
      hsh[tid] = h;
      int row = dir ? (SEQ - 1 - t) : t;
      out[(size_t)row * (2 * HID) + dir * HID + tid] = f2bf(h);
    }
    __syncthreads();
  }
}

// ---------------------------------------------------------------------------
// Char LSTM: one word per thread, weights in LDS.
// feat[j] = sum_{t<L} h_t[j] * (h_t . att_w) + att_b   (== gram@attW.T+b)
// ---------------------------------------------------------------------------
__global__ void k_char(const int* __restrict__ chars, const int* __restrict__ clen,
                       const float* __restrict__ cemb,
                       const float* __restrict__ cWih, const float* __restrict__ cWhh,
                       const float* __restrict__ cb,
                       const float* __restrict__ attW, const float* __restrict__ attB,
                       float* __restrict__ out /* [CHID][SEQ] */) {
  __shared__ float sWih[4 * CHID * CEMB], sWhh[4 * CHID * CHID], sB[4 * CHID], sAtt[CEMB];
  for (int i = threadIdx.x; i < 4 * CHID * CEMB; i += blockDim.x) sWih[i] = cWih[i];
  for (int i = threadIdx.x; i < 4 * CHID * CHID; i += blockDim.x) sWhh[i] = cWhh[i];
  for (int i = threadIdx.x; i < 4 * CHID; i += blockDim.x) sB[i] = cb[i];
  for (int i = threadIdx.x; i < CEMB; i += blockDim.x) sAtt[i] = attW[i];
  __syncthreads();
  int s = blockIdx.x * blockDim.x + threadIdx.x;
  if (s >= SEQ) return;
  int L = clen[s];
  float sarr[MAXW], feat[CHID];
  for (int pass = 0; pass < 2; ++pass) {
    float h[CHID], c[CHID];
    for (int j = 0; j < CHID; ++j) { h[j] = 0.0f; c[j] = 0.0f; }
    if (pass) for (int j = 0; j < CHID; ++j) feat[j] = attB[0];
    for (int t = 0; t < MAXW; ++t) {
      float x[CEMB];
      int ci = chars[s * MAXW + t];
      for (int j = 0; j < CEMB; ++j) x[j] = cemb[ci * CEMB + j];
      float z[4 * CHID];
      for (int r = 0; r < 4 * CHID; ++r) {
        float a = sB[r];
        for (int k = 0; k < CEMB; ++k) a = fmaf(sWih[r * CEMB + k], x[k], a);
        for (int k = 0; k < CHID; ++k) a = fmaf(sWhh[r * CHID + k], h[k], a);
        z[r] = a;
      }
      for (int u = 0; u < CHID; ++u) {
        c[u] = sigm(z[CHID + u]) * c[u] + sigm(z[u]) * tanhf(z[2 * CHID + u]);
        h[u] = sigm(z[3 * CHID + u]) * tanhf(c[u]);
      }
      if (pass == 0) {
        float d = 0.0f;
        for (int j = 0; j < CHID; ++j) d = fmaf(h[j], sAtt[j], d);
        sarr[t] = (t < L) ? d : 0.0f;
      } else if (t < L) {
        for (int j = 0; j < CHID; ++j) feat[j] = fmaf(h[j], sarr[t], feat[j]);
      }
    }
  }
  for (int j = 0; j < CHID; ++j) out[j * SEQ + s] = feat[j];
}

// Gather P[t][h] = label_head_a[best_arcs[t]][h] (f32, augmented+padded).
__global__ void k_gatherP(const int* __restrict__ best, const float* __restrict__ lh,
                          float* __restrict__ P) {
  int idx = blockIdx.x * blockDim.x + threadIdx.x;
  if (idx >= SEQ * KA) return;
  int t = idx / KA, n = idx % KA;
  float v = 0.0f;
  if (n < HID)       v = lh[(size_t)best[t] * HID + n];
  else if (n == HID) v = 1.0f;
  P[idx] = v;
}

// ---------------------------------------------------------------------------
// Label biaffine: per (l, 32t x 16h) WMMA tiles of label_dep_a @ W[l].T,
// fused epilogue dots tile rows with P[t][h] via shfl_xor butterflies,
// writes one partial per (l,t,ntile) -> deterministic (no atomics).
// ---------------------------------------------------------------------------
__global__ void k_label_biaff(const unsigned* __restrict__ A /* label_dep_a pairs */,
                              const unsigned* __restrict__ Bl /* 50 packed W[l].T slabs */,
                              const float* __restrict__ P,
                              float* __restrict__ selPart /* [LABN][SEQ][NTL] */) {
  int lane  = threadIdx.x & 31;
  int wave  = threadIdx.x >> 5;
  int ntile = blockIdx.x * 4 + wave;
  if (ntile >= NTL) return;
  int l = blockIdx.z;
  int mbase = blockIdx.y << 5;
  int n15 = lane & 15, hi = lane >> 4;
  int n = (ntile << 4) + n15;
  const uv4* A4 = (const uv4*)A;
  const uv4* B4 = (const uv4*)(Bl + (size_t)l * (KA / 2) * NLP);
  size_t arow0 = (size_t)(mbase + n15) * (KA >> 1);
  size_t arow1 = (size_t)(mbase + 16 + n15) * (KA >> 1);
  v8f c0 = {}, c1 = {};
  for (int kb = 0; kb < KA; kb += 32) {
    int ak = (kb >> 1) + 4 * hi;
    Frag16 a0, a1, b;
    size_t i0 = (arow0 + ak) >> 2, i1 = (arow1 + ak) >> 2;
    a0.q[0] = A4[i0];  a0.q[1] = A4[i0 + 2];
    a1.q[0] = A4[i1];  a1.q[1] = A4[i1 + 2];
    size_t bb = ((size_t)(kb >> 5) * NLP + n) * 4 + hi;
    b.q[0] = B4[bb];   b.q[1] = B4[bb + 2];
    c0 = __builtin_amdgcn_wmma_f32_16x16x32_bf16(false, a0.v, false, b.v,
                                                 (short)0, c0, false, false);
    c1 = __builtin_amdgcn_wmma_f32_16x16x32_bf16(false, a1.v, false, b.v,
                                                 (short)0, c1, false, false);
  }
#pragma unroll
  for (int r = 0; r < 8; ++r) {
    int t0 = mbase + r + 8 * hi;
    float p0 = c0[r] * P[(size_t)t0 * KA + n];
    float p1 = c1[r] * P[(size_t)(t0 + 16) * KA + n];
    for (int off = 1; off < 16; off <<= 1) {
      p0 += __shfl_xor(p0, off, 32);
      p1 += __shfl_xor(p1, off, 32);
    }
    if (n15 == 0) {
      selPart[((size_t)l * SEQ + t0) * NTL + ntile] = p0;
      selPart[((size_t)l * SEQ + t0 + 16) * NTL + ntile] = p1;
    }
  }
}

// Row softmax of S[512,512] -> arc_scores (axis=1).
__global__ void k_softmax(const float* __restrict__ S, float* __restrict__ out) {
  __shared__ float red[256];
  int i = blockIdx.x, tid = threadIdx.x;
  float m = -INFINITY;
  for (int j = tid; j < SEQ; j += 256) m = fmaxf(m, S[(size_t)i * SEQ + j]);
  red[tid] = m; __syncthreads();
  for (int s = 128; s > 0; s >>= 1) { if (tid < s) red[tid] = fmaxf(red[tid], red[tid + s]); __syncthreads(); }
  m = red[0]; __syncthreads();
  float sum = 0.0f;
  for (int j = tid; j < SEQ; j += 256) sum += __expf(S[(size_t)i * SEQ + j] - m);
  red[tid] = sum; __syncthreads();
  for (int s = 128; s > 0; s >>= 1) { if (tid < s) red[tid] += red[tid + s]; __syncthreads(); }
  float inv = 1.0f / red[0];
  for (int j = tid; j < SEQ; j += 256)
    out[(size_t)i * SEQ + j] = __expf(S[(size_t)i * SEQ + j] - m) * inv;
}

// Reduce selPart over tiles + log_softmax over l -> label_scores [LABN][SEQ].
__global__ void k_labelfinal(const float* __restrict__ selPart, float* __restrict__ out) {
  int t = blockIdx.x * blockDim.x + threadIdx.x;
  if (t >= SEQ) return;
  float sel[LABN];
  float m = -INFINITY;
  for (int l = 0; l < LABN; ++l) {
    const float* p = selPart + ((size_t)l * SEQ + t) * NTL;
    float a = 0.0f;
    for (int q = 0; q < NTL; ++q) a += p[q];
    sel[l] = a; m = fmaxf(m, a);
  }
  float sum = 0.0f;
  for (int l = 0; l < LABN; ++l) sum += __expf(sel[l] - m);
  float lse = m + logf(sum);
  for (int l = 0; l < LABN; ++l) out[(size_t)l * SEQ + t] = sel[l] - lse;
}

// ---------------------------------------------------------------------------
extern "C" void kernel_launch(void* const* d_in, const int* in_sizes, int n_in,
                              void* d_out, int out_size, void* d_ws, size_t ws_size,
                              hipStream_t stream) {
  (void)in_sizes; (void)n_in; (void)out_size; (void)ws_size;
  const int*   sentence = (const int*)d_in[0];
  const int*   tags     = (const int*)d_in[1];
  const int*   chars    = (const int*)d_in[2];
  const int*   clen     = (const int*)d_in[3];
  const int*   best     = (const int*)d_in[4];
  const float* wemb = (const float*)d_in[5];
  const float* temb = (const float*)d_in[6];
  const float* cemb = (const float*)d_in[7];
  const float* attW = (const float*)d_in[8];
  const float* attB = (const float*)d_in[9];
  const float* WihF = (const float*)d_in[10];
  const float* WhhF = (const float*)d_in[11];
  const float* bF   = (const float*)d_in[12];
  const float* WihB = (const float*)d_in[13];
  const float* WhhB = (const float*)d_in[14];
  const float* bB   = (const float*)d_in[15];
  const float* cWih = (const float*)d_in[16];
  const float* cWhh = (const float*)d_in[17];
  const float* cb   = (const float*)d_in[18];
  const float* adW = (const float*)d_in[19]; const float* adB = (const float*)d_in[20];
  const float* ahW = (const float*)d_in[21]; const float* ahB = (const float*)d_in[22];
  const float* ldW = (const float*)d_in[23]; const float* ldB = (const float*)d_in[24];
  const float* lhW = (const float*)d_in[25]; const float* lhB = (const float*)d_in[26];
  const float* biaffA = (const float*)d_in[27];
  const float* biaffL = (const float*)d_in[28];
  float* outArc  = (float*)d_out;            // [512][512]
  float* outLab  = outArc + SEQ * SEQ;       // [50][512]
  float* outChar = outLab + LABN * SEQ;      // [20][512]

  // workspace bump allocator (256B aligned)
  size_t off = 0;
  char* base = (char*)d_ws;
  auto alloc = [&](size_t bytes) -> void* {
    void* p = base + off;
    off += (bytes + 255) & ~(size_t)255;
    return p;
  };
  unsigned short* embA  = (unsigned short*)alloc((size_t)SEQ * KE * 2);
  unsigned short* embR  = (unsigned short*)alloc((size_t)SEQ * KE * 2);
  unsigned*       pWihF = (unsigned*)alloc((size_t)(KE / 2) * 2048 * 4);
  unsigned*       pWihB = (unsigned*)alloc((size_t)(KE / 2) * 2048 * 4);
  float*          ZXf   = (float*)alloc((size_t)SEQ * 2048 * 4);
  float*          ZXb   = (float*)alloc((size_t)SEQ * 2048 * 4);
  unsigned short* WhhFb = (unsigned short*)alloc((size_t)2048 * HID * 2);
  unsigned short* WhhBb = (unsigned short*)alloc((size_t)2048 * HID * 2);
  unsigned short* lstmO = (unsigned short*)alloc((size_t)SEQ * 2 * HID * 2);
  unsigned* pAD = (unsigned*)alloc((size_t)512 * 512 * 4);
  unsigned* pAH = (unsigned*)alloc((size_t)512 * 512 * 4);
  unsigned* pLD = (unsigned*)alloc((size_t)512 * 512 * 4);
  unsigned* pLH = (unsigned*)alloc((size_t)512 * 512 * 4);
  float*          arcDep  = (float*)alloc((size_t)SEQ * HID * 4);
  float*          labHead = (float*)alloc((size_t)SEQ * HID * 4);
  unsigned short* arcHA   = (unsigned short*)alloc((size_t)SEQ * KA * 2);
  unsigned short* labDA   = (unsigned short*)alloc((size_t)SEQ * KA * 2);
  unsigned*       pBArc   = (unsigned*)alloc((size_t)(KA / 2) * 512 * 4);
  unsigned short* Ubf     = (unsigned short*)alloc((size_t)SEQ * HID * 2);
  unsigned*       pADT    = (unsigned*)alloc((size_t)(HID / 2) * 512 * 4);
  float*          Sbuf    = (float*)alloc((size_t)SEQ * SEQ * 4);
  float*          Pbuf    = (float*)alloc((size_t)SEQ * KA * 4);
  unsigned*       pBLab   = (unsigned*)alloc((size_t)LABN * (KA / 2) * NLP * 4);
  float*          selPart = (float*)alloc((size_t)LABN * SEQ * NTL * 4);

  auto cdiv = [](int a, int b) { return (a + b - 1) / b; };

  // 1. embeddings + char LSTM (independent output)
  k_embed<<<cdiv(SEQ * KE, 256), 256, 0, stream>>>(sentence, tags, wemb, temb, embA, embR);
  k_char<<<2, 256, 0, stream>>>(chars, clen, cemb, cWih, cWhh, cb, attW, attB, outChar);

  // 2. weight packing
  k_packB<<<cdiv((KE / 2) * 2048, 256), 256, 0, stream>>>(WihF, 2048, 400, pWihF, KE, 2048, 0, 0);
  k_packB<<<cdiv((KE / 2) * 2048, 256), 256, 0, stream>>>(WihB, 2048, 400, pWihB, KE, 2048, 0, 0);
  k_convert<<<cdiv(2048 * HID, 256), 256, 0, stream>>>(WhhF, WhhFb, 2048 * HID);
  k_convert<<<cdiv(2048 * HID, 256), 256, 0, stream>>>(WhhB, WhhBb, 2048 * HID);
  k_packB<<<cdiv(512 * 512, 256), 256, 0, stream>>>(adW, 512, 1024, pAD, 1024, 512, 0, 0);
  k_packB<<<cdiv(512 * 512, 256), 256, 0, stream>>>(ahW, 512, 1024, pAH, 1024, 512, 0, 0);
  k_packB<<<cdiv(512 * 512, 256), 256, 0, stream>>>(ldW, 512, 1024, pLD, 1024, 512, 0, 0);
  k_packB<<<cdiv(512 * 512, 256), 256, 0, stream>>>(lhW, 512, 1024, pLH, 1024, 512, 0, 0);
  k_packB<<<cdiv((KA / 2) * 512, 256), 256, 0, stream>>>(biaffA, 512, 513, pBArc, KA, 512, 0, 0);
  {
    dim3 g(cdiv((KA / 2) * NLP, 256), 1, LABN);
    k_packB<<<g, 256, 0, stream>>>(biaffL, 513, 513, pBLab, KA, NLP,
                                   (long)513 * 513, (long)(KA / 2) * NLP);
  }
  k_init_aug<<<cdiv(2 * SEQ * 32, 256), 256, 0, stream>>>(arcHA, labDA);

  // 3. input projections (WMMA) then the two recurrences (resident blocks)
  {
    dim3 g(2048 / 16 / 4, SEQ / 32);
    k_gemm<true, false, true, false><<<g, 128, 0, stream>>>(
        (const unsigned*)embA, KE, pWihF, 2048, ZXf, 2048, nullptr, 0, bF);
    k_gemm<true, false, true, false><<<g, 128, 0, stream>>>(
        (const unsigned*)embR, KE, pWihB, 2048, ZXb, 2048, nullptr, 0, bB);
  }
  k_lstm<<<2, 1024, 0, stream>>>(ZXf, ZXb, WhhFb, WhhBb, lstmO);

  // 4. four MLPs (WMMA, fused bias+relu, mixed f32/bf16 epilogues)
  {
    dim3 g(512 / 16 / 4, SEQ / 32);
    k_gemm<true, false, true, true><<<g, 128, 0, stream>>>(
        (const unsigned*)lstmO, 1024, pAD, 512, arcDep, 512, nullptr, 0, adB);
    k_gemm<false, true, true, true><<<g, 128, 0, stream>>>(
        (const unsigned*)lstmO, 1024, pAH, 512, nullptr, 0, arcHA, KA, ahB);
    k_gemm<false, true, true, true><<<g, 128, 0, stream>>>(
        (const unsigned*)lstmO, 1024, pLD, 512, nullptr, 0, labDA, KA, ldB);
    k_gemm<true, false, true, true><<<g, 128, 0, stream>>>(
        (const unsigned*)lstmO, 1024, pLH, 512, labHead, 512, nullptr, 0, lhB);
  }

  // 5. arc biaffine: U = arc_head_a @ biaff_arc_W.T ; S = U @ arc_dep.T ; softmax
  {
    dim3 g(512 / 16 / 4, SEQ / 32);
    k_gemm<false, true, false, false><<<g, 128, 0, stream>>>(
        (const unsigned*)arcHA, KA, pBArc, 512, nullptr, 0, Ubf, 512, nullptr);
    k_packB<<<cdiv((HID / 2) * 512, 256), 256, 0, stream>>>(arcDep, 512, 512, pADT, 512, 512, 0, 0);
    k_gemm<true, false, false, false><<<g, 128, 0, stream>>>(
        (const unsigned*)Ubf, 512, pADT, 512, Sbuf, 512, nullptr, 0, nullptr);
  }
  k_softmax<<<SEQ, 256, 0, stream>>>(Sbuf, outArc);

  // 6. label biaffine (batched WMMA + fused selection dot) and log_softmax
  k_gatherP<<<cdiv(SEQ * KA, 256), 256, 0, stream>>>(best, labHead, Pbuf);
  {
    dim3 g(cdiv(NTL, 4), SEQ / 32, LABN);
    k_label_biaff<<<g, 128, 0, stream>>>((const unsigned*)labDA, pBLab, Pbuf, selPart);
  }
  k_labelfinal<<<cdiv(SEQ, 256), 256, 0, stream>>>(selPart, outLab);
}
